// LocationAwareAttention_45595372814748
// MI455X (gfx1250) — matrix operand order repaired and checked by reading, not compiled, not run on hardware
//
#include <hip/hip_runtime.h>
#include <hip/hip_bf16.h>

#define B_ 32
#define T_ 2048
#define H_ 512

typedef __bf16 v16bf __attribute__((ext_vector_type(16)));
typedef float  v8f   __attribute__((ext_vector_type(8)));

static __device__ __forceinline__ unsigned short f2bf(float f) {
    __bf16 h = (__bf16)f;            // RNE float->bf16
    unsigned short u;
    __builtin_memcpy(&u, &h, sizeof(u));
    return u;
}

// ---------------------------------------------------------------------------
// Convert v_w (H x H fp32, row-major [o][k]) into a pre-swizzled bf16 buffer
// in the per-lane WMMA B-matrix (32x16 bf16, K x N) order k_score uses (contiguous k):
//   bw[((ntile*16 + kt)*32 + lane)*16 + s]  holds  v_w[ntile*16 + (lane&15),
//                                                      kt*32 + (lane>=16?16:0) + s]
// so each lane later loads its 16-half fragment with two 16-byte loads.
// ---------------------------------------------------------------------------
__global__ void k_prep_b(const float* __restrict__ v_w, unsigned short* __restrict__ bw) {
    int idx = blockIdx.x * blockDim.x + threadIdx.x;   // [0, 512*512)
    int ntile = idx >> 13;            // 16*32*16 = 8192 slots per n-tile
    int rem   = idx & 8191;
    int kt    = rem >> 9;
    int rem2  = rem & 511;
    int lane  = rem2 >> 4;
    int s     = rem2 & 15;
    int n = ntile * 16 + (lane & 15);
    int k = kt * 32 + ((lane >> 4) << 4) + s;
    bw[idx] = f2bf(v_w[n * H_ + k]);
}

// qb[b][o] = sum_h query[b][h] * q_w[o][h] + bias[o]   (tiny GEMV, fp32)
__global__ void k_prep_qb(const float* __restrict__ query, const float* __restrict__ q_w,
                          const float* __restrict__ bias, float* __restrict__ qb) {
    int idx = blockIdx.x * blockDim.x + threadIdx.x;   // [0, B*H)
    int b = idx >> 9, o = idx & (H_ - 1);
    const float* qrow = query + b * H_;
    const float* wrow = q_w + o * H_;
    float s = bias[o];
    for (int h = 0; h < H_; ++h) s += qrow[h] * wrow[h];
    qb[idx] = s;
}

// ---------------------------------------------------------------------------
// Fused: scores[b,t] = sigmoid( s_w . tanh(value[b,t]@v_w^T + qb[b] + conv) + s_b )
// One WG = 256 threads (8 wave32) handles 32 rows (b fixed) x all 512 channels.
// Wave w owns n-tiles [4w, 4w+4); M-tiles = 2 (rows 0-15, 16-31).
// kt is the outer loop so A fragments are loaded once per K-step and shared
// across all 4 n-tile accumulators (8 WMMA per 4 ds_load_b128 + 8 global_b128).
// ---------------------------------------------------------------------------
__global__ void __launch_bounds__(256) k_score(
        const float* __restrict__ value, const float* __restrict__ last_attn,
        const float* __restrict__ conv_w, const float* __restrict__ conv_b,
        const float* __restrict__ s_w, const float* __restrict__ s_b,
        const float* __restrict__ qb, const unsigned short* __restrict__ bw,
        float* __restrict__ scores) {
    __shared__ unsigned short aLds[32 * H_];   // 32 rows x 512 k, bf16 (32 KB)
    __shared__ float laLds[34];                // last_attn[t0-1 .. t0+32]
    __shared__ float wavePart[8 * 32];         // per-wave per-row partial sums

    const int tid    = threadIdx.x;
    const int lane   = tid & 31;
    const int w      = tid >> 5;
    const int laneHi = (lane >> 4) & 1;
    const int lane15 = lane & 15;
    const int b  = blockIdx.x >> 6;            // 64 tiles of 32 rows per batch
    const int t0 = (blockIdx.x & 63) << 5;

    // ---- stage value tile as bf16 into LDS (row-major [row][k]) ----
    const float4* vsrc = (const float4*)value;
    for (int i = 0; i < 16; ++i) {
        int e   = tid + i * 256;               // float4 index in tile [0,4096)
        int row = e >> 7;                      // 128 float4 per row
        int c4  = e & 127;
        float4 v = vsrc[(size_t)(b * T_ + t0 + row) * (H_ / 4) + c4];
        ushort4 p;
        p.x = f2bf(v.x); p.y = f2bf(v.y); p.z = f2bf(v.z); p.w = f2bf(v.w);
        *(ushort4*)&aLds[row * H_ + c4 * 4] = p;
    }
    if (tid < 34) {
        int t = t0 - 1 + tid;
        laLds[tid] = (t >= 0 && t < T_) ? last_attn[b * T_ + t] : 0.0f;
    }
    __syncthreads();

    v8f acc[2][4];                             // [mtile][ntile-local]
#pragma unroll
    for (int mt = 0; mt < 2; ++mt)
#pragma unroll
        for (int nt = 0; nt < 4; ++nt) acc[mt][nt] = (v8f){0,0,0,0,0,0,0,0};

#pragma unroll 2
    for (int kt = 0; kt < 16; ++kt) {
        // A fragments from LDS in contiguous k order (same as bw):
        // lane<16: K {kb..kb+15}; lane>=16: K {kb+16..kb+31}
        const int kbase = kt * 32 + laneHi * 16;
        v16bf afrag[2];
#pragma unroll
        for (int mt = 0; mt < 2; ++mt) {
            const unsigned short* ap = &aLds[(mt * 16 + lane15) * H_ + kbase];
            ((uint4*)&afrag[mt])[0] = *(const uint4*)(ap);
            ((uint4*)&afrag[mt])[1] = *(const uint4*)(ap + 8);
        }
#pragma unroll
        for (int nt = 0; nt < 4; ++nt) {
            const int ntg = w * 4 + nt;
            // B fragment: pre-swizzled, 32 contiguous bytes per lane (L2-resident)
            v16bf bfrag;
            const unsigned short* bp = bw + ((size_t)(ntg * 16 + kt) * 32 + lane) * 16;
            ((uint4*)&bfrag)[0] = *(const uint4*)(bp);
            ((uint4*)&bfrag)[1] = *(const uint4*)(bp + 8);
#pragma unroll
            for (int mt = 0; mt < 2; ++mt)
                acc[mt][nt] = __builtin_amdgcn_wmma_f32_16x16x32_bf16(
                    false, afrag[mt], false, bfrag, (short)0, acc[mt][nt], false, false);
        }
    }

    // ---- epilogue: add qb + conv, tanh, weight by s_w, reduce over channels ----
    float rowAcc[2][8];
#pragma unroll
    for (int mt = 0; mt < 2; ++mt)
#pragma unroll
        for (int r = 0; r < 8; ++r) rowAcc[mt][r] = 0.0f;

#pragma unroll
    for (int nt = 0; nt < 4; ++nt) {
        const int n  = (w * 4 + nt) * 16 + lane15;
        const float qbn = qb[b * H_ + n];
        const float cw0 = conv_w[n * 3 + 0];
        const float cw1 = conv_w[n * 3 + 1];
        const float cw2 = conv_w[n * 3 + 2];
        const float cbn = conv_b[n];
        const float swn = s_w[n];
#pragma unroll
        for (int mt = 0; mt < 2; ++mt)
#pragma unroll
            for (int r = 0; r < 8; ++r) {
                int lrow = mt * 16 + laneHi * 8 + r;   // C/D: row = r + (laneHi?8:0)
                float conv = cw0 * laLds[lrow] + cw1 * laLds[lrow + 1]
                           + cw2 * laLds[lrow + 2] + cbn;
                float e = acc[mt][nt][r] + qbn + conv;
                rowAcc[mt][r] += tanhf(e) * swn;
            }
    }

    // reduce over the 16 lanes (channels) of each half-wave group
#pragma unroll
    for (int mt = 0; mt < 2; ++mt)
#pragma unroll
        for (int r = 0; r < 8; ++r)
#pragma unroll
            for (int off = 1; off < 16; off <<= 1)
                rowAcc[mt][r] += __shfl_xor(rowAcc[mt][r], off, 32);

    if (lane15 == 0) {
#pragma unroll
        for (int mt = 0; mt < 2; ++mt)
#pragma unroll
            for (int r = 0; r < 8; ++r)
                wavePart[w * 32 + mt * 16 + laneHi * 8 + r] = rowAcc[mt][r];
    }
    __syncthreads();

    if (tid < 32) {  // fixed-order cross-wave sum -> deterministic
        float s = 0.0f;
#pragma unroll
        for (int wv = 0; wv < 8; ++wv) s += wavePart[wv * 32 + tid];
        s += s_b[0];
        scores[b * T_ + t0 + tid] = 1.0f / (1.0f + expf(-s));
    }
}

// per-batch L1 normalize -> attn output
__global__ void k_norm(const float* __restrict__ scores, float* __restrict__ attn) {
    __shared__ float red[256];
    int b = blockIdx.x, tid = threadIdx.x;
    float s = 0.0f;
    for (int i = tid; i < T_; i += 256) s += scores[b * T_ + i];
    red[tid] = s;
    __syncthreads();
    for (int o = 128; o > 0; o >>= 1) {
        if (tid < o) red[tid] += red[tid + o];
        __syncthreads();
    }
    float inv = 1.0f / red[0];
    for (int i = tid; i < T_; i += 256) attn[b * T_ + i] = scores[b * T_ + i] * inv;
}

// context partials: block (b, tc); each half-block (128 threads = all 128
// float4 channel-groups) accumulates 128 t-rows with b128 loads.
// part layout: [16 partials][B][H], partial p = tc*2 + half. Deterministic.
__global__ void k_ctx_part(const float* __restrict__ value, const float* __restrict__ attn,
                           float* __restrict__ part) {
    __shared__ float sa[256];
    int b = blockIdx.x >> 3, tc = blockIdx.x & 7, tid = threadIdx.x;
    sa[tid] = attn[b * T_ + tc * 256 + tid];
    __syncthreads();
    int half = tid >> 7;
    int g    = tid & 127;                       // float4 channel group [0,128)
    float4 acc = {0.0f, 0.0f, 0.0f, 0.0f};
    const float4* v4 = (const float4*)value;
    for (int tt = 0; tt < 128; ++tt) {
        int rloc = half * 128 + tt;
        float av = sa[rloc];
        float4 v = v4[(size_t)(b * T_ + tc * 256 + rloc) * (H_ / 4) + g];
        acc.x += av * v.x; acc.y += av * v.y; acc.z += av * v.z; acc.w += av * v.w;
    }
    float4* p4 = (float4*)part;
    p4[(size_t)(tc * 2 + half) * (B_ * H_ / 4) + b * (H_ / 4) + g] = acc;
}

__global__ void k_ctx_reduce(const float* __restrict__ part, float* __restrict__ ctx) {
    int i = blockIdx.x * blockDim.x + threadIdx.x;   // [0, B*H)
    float s = 0.0f;
#pragma unroll
    for (int p = 0; p < 16; ++p) s += part[p * (B_ * H_) + i];
    ctx[i] = s;
}

extern "C" void kernel_launch(void* const* d_in, const int* in_sizes, int n_in,
                              void* d_out, int out_size, void* d_ws, size_t ws_size,
                              hipStream_t stream) {
    const float* query     = (const float*)d_in[0];  // (B,1,H)
    const float* value     = (const float*)d_in[1];  // (B,T,H)
    const float* last_attn = (const float*)d_in[2];  // (B,T)
    const float* conv_w    = (const float*)d_in[3];  // (H,1,3)
    const float* conv_b    = (const float*)d_in[4];  // (H)
    const float* q_w       = (const float*)d_in[5];  // (H,H)
    const float* v_w       = (const float*)d_in[6];  // (H,H)
    const float* s_w       = (const float*)d_in[7];  // (1,H)
    const float* s_b       = (const float*)d_in[8];  // (1)
    const float* bias      = (const float*)d_in[9];  // (H)
    (void)in_sizes; (void)n_in; (void)out_size; (void)ws_size;

    float* ctx_out  = (float*)d_out;            // context: B*H floats
    float* attn_out = (float*)d_out + B_ * H_;  // attn:    B*T floats

    char* ws = (char*)d_ws;
    unsigned short* bw = (unsigned short*)ws;                           // 512 KB
    float* qb     = (float*)(ws + 512 * 1024);                          //  64 KB
    float* scores = (float*)(ws + 512 * 1024 + 64 * 1024);              // 256 KB
    float* part   = (float*)(ws + 512 * 1024 + 64 * 1024 + 256 * 1024); //   1 MB

    k_prep_b    <<<(H_ * H_) / 256, 256, 0, stream>>>(v_w, bw);
    k_prep_qb   <<<(B_ * H_) / 256, 256, 0, stream>>>(query, q_w, bias, qb);
    k_score     <<<(B_ * T_) / 32, 256, 0, stream>>>(value, last_attn, conv_w, conv_b,
                                                     s_w, s_b, qb, bw, scores);
    k_norm      <<<B_, 256, 0, stream>>>(scores, attn_out);
    k_ctx_part  <<<B_ * 8, 256, 0, stream>>>(value, attn_out, part);
    k_ctx_reduce<<<(B_ * H_) / 256, 256, 0, stream>>>(part, ctx_out);
}
